// Event_SMLP_35381940585135
// MI455X (gfx1250) — matrix operand order repaired and checked by reading, not compiled, not run on hardware
//
#include <hip/hip_runtime.h>
#include <stdint.h>

typedef _Float16 v8h  __attribute__((ext_vector_type(8)));
typedef _Float16 v16h __attribute__((ext_vector_type(16)));
typedef float    v8f  __attribute__((ext_vector_type(8)));

#define B_TOT   8192
#define IN_DIM  784
#define IN_PAD  800      // 25 K-tiles of 32
#define HID     400
#define HID_PAD 512      // 32 N-tiles of 16 (8 waves x 4 tiles)
#define OUT_N   10
#define BT      64       // batch rows per workgroup (4 WMMA M-tiles)
#define PPT     98       // spike pairs per thread per step (4 threads/row)

// ---------------------------------------------------------------- utilities
__device__ __forceinline__ uint32_t pcg_hash(uint32_t v) {
  v = v * 747796405u + 2891336453u;
  uint32_t w = ((v >> ((v >> 28u) + 4u)) ^ v) * 277803737u;
  return (w >> 22u) ^ w;
}

__device__ __forceinline__ v16h cat16(v8h lo, v8h hi) {
  return __builtin_shufflevector(lo, hi, 0,1,2,3,4,5,6,7,8,9,10,11,12,13,14,15);
}

__device__ __forceinline__ v8f wmma_f16(v16h a, v16h b, v8f c) {
  return __builtin_amdgcn_wmma_f32_16x16x32_f16(false, a, false, b, (short)0, c,
                                                false, false);
}

// two Bernoulli spikes from one hash; strength-reduced addressing:
// ip/xrow are per-thread loop-invariant row pointers, fidx is the pair slot
__device__ __forceinline__ void gen2(const float* __restrict__ ip,
                                     _Float16* __restrict__ xrow,
                                     uint32_t sbase, int fidx) {
  uint32_t h = pcg_hash(sbase + (uint32_t)fidx);
  float p0 = ip[2 * fidx];
  float p1 = ip[2 * fidx + 1];
  uint32_t s0 = (p0 > (float)(h & 0xFFFFu) * 0x1p-16f) ? 0x3C00u : 0u;
  uint32_t s1 = (p1 > (float)(h >> 16) * 0x1p-16f) ? 0x3C00u : 0u;
  *(uint32_t*)(xrow + 2 * fidx) = s0 | (s1 << 16);   // 4B aligned
}

// ------------------------------------------------- weight conversion kernels
__global__ __launch_bounds__(256)
void cvt_w1(const float* __restrict__ w1, _Float16* __restrict__ w1c) {
  int i = blockIdx.x * 256 + threadIdx.x;           // over HID_PAD * IN_PAD
  if (i >= HID_PAD * IN_PAD) return;
  int n = i / IN_PAD, k = i - n * IN_PAD;
  float v = (n < HID && k < IN_DIM) ? w1[n * IN_DIM + k] : 0.0f;
  w1c[i] = (_Float16)v;
}

__global__ __launch_bounds__(256)
void cvt_w2(const float* __restrict__ w2, _Float16* __restrict__ w2c) {
  int i = blockIdx.x * 256 + threadIdx.x;           // over 16 * HID_PAD
  if (i >= 16 * HID_PAD) return;
  int n = i / HID_PAD, k = i - n * HID_PAD;
  float v = (n < OUT_N && k < HID) ? w2[n * HID + k] : 0.0f;
  w2c[i] = (_Float16)v;
}

// ------------------------------------------------------------- main SNN kernel
__global__ __launch_bounds__(256, 1)
void snn_persistent(const float* __restrict__ input,
                    const _Float16* __restrict__ w1c,
                    const _Float16* __restrict__ w2c,
                    const int* __restrict__ twp,
                    float* __restrict__ out) {
  // 270 KB static LDS (<= 320 KB/WGP): double-buffered spikes + s1
  __shared__ _Float16 xbuf0[BT][IN_PAD];
  __shared__ _Float16 xbuf1[BT][IN_PAD];
  __shared__ _Float16 s1buf[BT][HID_PAD];

  const int tid  = threadIdx.x;
  const int lane = tid & 31;
  const int wv   = tid >> 5;      // wave id 0..7
  const int half = lane >> 4;     // 0: lanes 0-15, 1: lanes 16-31
  const int lm   = lane & 15;     // m for A-tiles / n for B- and C-tiles
  const int b0   = blockIdx.x * BT;
  const int T    = *twp;

  // fixed fill assignment: thread owns row fm, pair slots [0, PPT) at fk0
  const int fm  = tid >> 2;              // 0..63
  const int fk0 = (tid & 3) * PPT;       // 0,98,196,294 (pairs; *2 = k)
  const float* fip = input + (b0 + fm) * IN_DIM + fk0 * 2;
  const uint32_t fsb = (uint32_t)((b0 + fm) * (IN_DIM / 2) + fk0);

  // one-time: zero the K-padding columns of both spike buffers
  for (int i = tid; i < BT * (IN_PAD - IN_DIM); i += 256) {
    int m = i >> 4, k = IN_DIM + (i & 15);
    xbuf0[m][k] = (_Float16)0.0f;
    xbuf1[m][k] = (_Float16)0.0f;
  }
  // prologue: spikes for t = 0
  {
    _Float16* xrow = &xbuf0[fm][fk0 * 2];
    #pragma unroll 2
    for (int f = 0; f < PPT; ++f) gen2(fip, xrow, fsb, f);
  }

  // persistent membrane state in WMMA accumulators
  v8f h1[4][4] = {};         // [Mtile][local Ntile], 128 VGPRs
  v8f h2   = {};             // layer-2 membrane (waves 0-3, Mtile = wv)
  v8f accs = {};             // output spike accumulator (waves 0-3)

  for (int t = 0; t < T; ++t) {
    _Float16 (*xcur)[IN_PAD] = (t & 1) ? xbuf1 : xbuf0;
    _Float16 (*xnxt)[IN_PAD] = (t & 1) ? xbuf0 : xbuf1;
    const bool do_fill = (t + 1 < T);
    _Float16* fxrow = &xnxt[fm][fk0 * 2];
    const uint32_t fseed = (uint32_t)(t + 1) * 0x9E3779B9u + fsb;

    __syncthreads();  // xcur writes visible; prior readers of xnxt/s1 done

    // ---- layer 1 GEMM (XDL) with next-step spike generation (VALU)
    //      interleaved -- 16-bit WMMA co-executes with VALU on CDNA5 ----
    #pragma unroll 1
    for (int kt = 0; kt < IN_PAD / 32; ++kt) {
      const int kA = kt * 32 + half * 8;   // ISA 16-bit A layout
      v16h A[4];
      #pragma unroll
      for (int mi = 0; mi < 4; ++mi) {
        const int row = mi * 16 + lm;
        A[mi] = cat16(*(const v8h*)&xcur[row][kA],
                      *(const v8h*)&xcur[row][kA + 16]);
      }
      #pragma unroll
      for (int j = 0; j < 4; ++j) {
        const int nrow = (wv * 4 + j) * 16 + lm;   // hidden unit index
        const _Float16* bp = w1c + nrow * IN_PAD + kt * 32 + half * 8;
        if (j == 0) __builtin_prefetch(bp + 32, 0, 3);  // next K chunk
        v16h Bm = cat16(*(const v8h*)bp, *(const v8h*)(bp + 16));
        #pragma unroll
        for (int mi = 0; mi < 4; ++mi)
          h1[mi][j] = wmma_f16(A[mi], Bm, h1[mi][j]);   // B reused across 4 M
      }
      // 4 spike pairs for step t+1 per thread per K-tile (rides VALU pipe);
      // tail guard is scalar-uniform (kt, f uniform) -> no EXEC churn
      if (do_fill) {
        #pragma unroll
        for (int f = 0; f < 4; ++f) {
          const int fidx = kt * 4 + f;
          if (fidx < PPT) gen2(fip, fxrow, fseed, fidx);
        }
      }
    }

    // ---- threshold / reset / decay; export s1 spikes (f16 bit select) ----
    #pragma unroll
    for (int j = 0; j < 4; ++j) {
      const int ng = (wv * 4 + j) * 16 + lm;   // C-layout: lane -> column n
      #pragma unroll
      for (int mi = 0; mi < 4; ++mi) {
        #pragma unroll
        for (int r = 0; r < 8; ++r) {
          const int m = mi * 16 + half * 8 + r;  // C-layout: VGPR r -> row m
          float v = h1[mi][j][r];
          const bool fire = v > 0.5f;
          *(uint16_t*)&s1buf[m][ng] = fire ? (uint16_t)0x3C00u : (uint16_t)0u;
          h1[mi][j][r] = fire ? 0.0f : v * 0.2f;
        }
      }
    }
    __syncthreads();

    // ---- layer 2 on waves 0-3 (one M-tile each): h2 += S1 @ W2^T ----
    if (wv < 4) {
      #pragma unroll 1
      for (int kt = 0; kt < HID_PAD / 32; ++kt) {
        const int kA = kt * 32 + half * 8;
        const int row = wv * 16 + lm;
        v16h A = cat16(*(const v8h*)&s1buf[row][kA],
                       *(const v8h*)&s1buf[row][kA + 16]);
        const _Float16* bp = w2c + lm * HID_PAD + kt * 32 + half * 8;
        v16h Bm = cat16(*(const v8h*)bp, *(const v8h*)(bp + 16));
        h2 = wmma_f16(A, Bm, h2);
      }
      #pragma unroll
      for (int r = 0; r < 8; ++r) {
        float v = h2[r];
        const bool fire = v > 0.5f;
        accs[r] += fire ? 1.0f : 0.0f;
        h2[r] = fire ? 0.0f : v * 0.2f;
      }
    }
  }

  // ---- write averaged output spikes ----
  if (wv < 4 && lm < OUT_N) {
    const float inv = 1.0f / (float)T;
    #pragma unroll
    for (int r = 0; r < 8; ++r) {
      const int m = wv * 16 + half * 8 + r;
      out[(b0 + m) * OUT_N + lm] = accs[r] * inv;
    }
  }
}

// ---------------------------------------------------------------------- host
extern "C" void kernel_launch(void* const* d_in, const int* in_sizes, int n_in,
                              void* d_out, int out_size, void* d_ws, size_t ws_size,
                              hipStream_t stream) {
  const float* input = (const float*)d_in[0];
  const float* w1    = (const float*)d_in[1];
  const float* w2    = (const float*)d_in[2];
  const int*   tw    = (const int*)d_in[3];
  float* out = (float*)d_out;

  _Float16* w1c = (_Float16*)d_ws;
  _Float16* w2c = (_Float16*)((char*)d_ws +
                              (size_t)HID_PAD * IN_PAD * sizeof(_Float16));

  cvt_w1<<<(HID_PAD * IN_PAD + 255) / 256, 256, 0, stream>>>(w1, w1c);
  cvt_w2<<<(16 * HID_PAD + 255) / 256, 256, 0, stream>>>(w2, w2c);
  snn_persistent<<<B_TOT / BT, 256, 0, stream>>>(input, w1c, w2c, tw, out);
}
